// CondRotConv2d_32083405701664
// MI455X (gfx1250) — matrix operand order repaired and checked by reading, not compiled, not run on hardware
//
#include <hip/hip_runtime.h>
#include <hip/hip_bf16.h>

typedef __attribute__((ext_vector_type(16))) __bf16 v16bf;
typedef __attribute__((ext_vector_type(8)))  float  v8f;
typedef __attribute__((ext_vector_type(4)))  float  v4f;

#define CIN   32
#define COUT  32
#define EXP   8
#define HH    256
#define WW    256
#define BB    16
#define KTOT  288   // CIN * 9

// ---------------------------------------------------------------------------
// Stage 1: per-(b,ci) mean over HxW.  One block per (b,ci).
// Also warms L2 with x (134 MB < 192 MB L2) for the conv pass.
// ---------------------------------------------------------------------------
__global__ __launch_bounds__(256) void avg_kernel(const float* __restrict__ x,
                                                  float* __restrict__ ws_avg) {
    const int bc  = blockIdx.x;              // 0..511 = b*32+ci
    const int tid = threadIdx.x;
    const v4f* p  = (const v4f*)(x + (size_t)bc * (HH * WW));
    float s = 0.0f;
    #pragma unroll 4
    for (int i = tid; i < (HH * WW) / 4; i += 256) {
        v4f v = p[i];
        s += v.x + v.y + v.z + v.w;
    }
    __shared__ float red[256];
    red[tid] = s;
    __syncthreads();
    for (int off = 128; off > 0; off >>= 1) {
        if (tid < off) red[tid] += red[tid + off];
        __syncthreads();
    }
    if (tid == 0) ws_avg[bc] = red[0] * (1.0f / (HH * WW));
}

// ---------------------------------------------------------------------------
// Stage 2: gate/theta -> gated rotation matrices -> per-sample bf16 weights,
// written directly in WMMA A-fragment order:
//   frag[ ((b*9 + t)*2 + r)*32 + lane ][ s ]  (16 bf16 per lane)
// with M = lane&15, K = t*32 + base + (s<8 ? s : 8+s), base = 8*(lane>=16).
// (ISA 7.12.2, 16-bit A-matrix 16x32)
// ---------------------------------------------------------------------------
__global__ __launch_bounds__(256) void wgen_kernel(const float* __restrict__ ws_avg,
                                                   const float* __restrict__ weight,
                                                   const float* __restrict__ fc_a_w,
                                                   const float* __restrict__ fc_a_b,
                                                   const float* __restrict__ fc_t_w,
                                                   const float* __restrict__ fc_t_b,
                                                   __bf16* __restrict__ frag) {
    const int b   = blockIdx.x;      // 0..15
    const int cog = blockIdx.y;      // 0..7  -> co in [cog*4, cog*4+4)
    const int tid = threadIdx.x;

    __shared__ float sg[EXP];            // gate
    __shared__ float sth[EXP];           // theta
    __shared__ float sAe[EXP * 81];      // gate[e] * M[e][k][j]
    __shared__ float sW[EXP * 4 * KTOT]; // weight slice [e][co_local][ci*9+j]

    if (tid < 16) {
        const int  e  = tid & 7;
        const bool th = tid >= 8;
        const float* wv = th ? fc_t_w : fc_a_w;
        float s = th ? fc_t_b[e] : fc_a_b[e];
        #pragma unroll
        for (int ci = 0; ci < CIN; ++ci)
            s += ws_avg[b * CIN + ci] * wv[e * CIN + ci];
        const float sig = 1.0f / (1.0f + __expf(-s));
        if (th) sth[e] = sig;    // PROPORTION == 1.0
        else    sg[e]  = sig;
    }

    for (int idx = tid; idx < EXP * 4 * KTOT; idx += 256) {
        const int e  = idx / (4 * KTOT);
        const int rm = idx - e * (4 * KTOT);
        const int cl = rm / KTOT;
        const int q  = rm - cl * KTOT;
        sW[idx] = weight[(size_t)(e * COUT + cog * 4 + cl) * KTOT + q];
    }
    __syncthreads();

    for (int idx = tid; idx < EXP * 81; idx += 256) {
        const int e  = idx / 81;
        const int rm = idx - e * 81;
        const int k  = rm / 9;
        const int j  = rm - k * 9;
        const float gr_k = (float)(k / 3 - 1), gc_k = (float)(k % 3 - 1);
        const float gr_j = (float)(j / 3 - 1), gc_j = (float)(j % 3 - 1);
        const float c = __cosf(sth[e]), s = __sinf(sth[e]);
        const float src_r = c * gr_k - s * gc_k;
        const float src_c = s * gr_k + c * gc_k;
        const float wr = fmaxf(0.0f, 1.0f - fabsf(src_r - gr_j));
        const float wc = fmaxf(0.0f, 1.0f - fabsf(src_c - gc_j));
        sAe[idx] = sg[e] * wr * wc;
    }
    __syncthreads();

    for (int idx = tid; idx < 4 * KTOT; idx += 256) {
        const int cl   = idx / KTOT;
        const int Kidx = idx - cl * KTOT;
        const int ci   = Kidx / 9;
        const int k    = Kidx - ci * 9;
        float acc = 0.0f;
        #pragma unroll
        for (int e = 0; e < EXP; ++e) {
            #pragma unroll
            for (int j = 0; j < 9; ++j)
                acc += sAe[e * 81 + k * 9 + j] * sW[(e * 4 + cl) * KTOT + ci * 9 + j];
        }
        const int co   = cog * 4 + cl;
        const int t    = Kidx >> 5;
        const int koff = Kidx & 31;
        const int r    = co >> 4;
        const int M    = co & 15;
        const int hi   = (koff >= 16);
        const int rem  = koff & 15;
        const int lg   = (rem >= 8);
        const int slow = rem - (lg ? 8 : 0);
        const int s    = hi ? (8 + slow) : slow;
        const int lane = M + (lg ? 16 : 0);
        frag[(size_t)((((b * 9 + t) * 2 + r) * 32 + lane)) * 16 + s] = (__bf16)acc;
    }
}

// ---------------------------------------------------------------------------
// Stage 3: implicit-GEMM 3x3 conv via V_WMMA_F32_16X16X32_BF16.
// Block = (b, h, 128-wide w tile); 8 waves = 8 spatial 16-subtiles.
// Each wave holds TWO accumulators (co 0-15 and 16-31) and reuses each
// B fragment for 2 WMMAs.  K = 288 = 9 WMMA K-tiles.
// Boundary handling is branch-free: clamped address + select-to-zero.
// ---------------------------------------------------------------------------
__global__ __launch_bounds__(256) void conv_kernel(const float* __restrict__ x,
                                                   const __bf16* __restrict__ frag,
                                                   float* __restrict__ out) {
    const int bx   = blockIdx.x;              // 16*256*2
    const int wt   = bx & 1;
    const int h    = (bx >> 1) & (HH - 1);
    const int b    = bx >> 9;
    const int lane = threadIdx.x & 31;
    const int pt   = threadIdx.x >> 5;        // 0..7
    const int w0   = wt * 128 + pt * 16;

    v8f acc0 = {0.f, 0.f, 0.f, 0.f, 0.f, 0.f, 0.f, 0.f};
    v8f acc1 = {0.f, 0.f, 0.f, 0.f, 0.f, 0.f, 0.f, 0.f};

    #pragma unroll
    for (int t = 0; t < 9; ++t) {
        // --- A fragments for both co-halves: one 32B load each ---
        const v16bf a0 =
            *(const v16bf*)(frag + (size_t)(((b * 9 + t) * 2 + 0) * 32 + lane) * 16);
        const v16bf a1 =
            *(const v16bf*)(frag + (size_t)(((b * 9 + t) * 2 + 1) * 32 + lane) * 16);

        // --- B fragment: lane -> column w0+(lane&15); element i -> K row (table order) ---
        v16bf bf;
        #pragma unroll
        for (int i = 0; i < 16; ++i) {
            const int kk  = (i < 8) ? (8 * (lane >> 4) + i) : (16 + 8 * (lane >> 4) + (i - 8));
            const int Kg  = t * 32 + kk;          // 0..287
            const int ci  = Kg / 9;
            const int tap = Kg - ci * 9;
            const int dr  = tap / 3 - 1;
            const int dc  = tap - (tap / 3) * 3 - 1;
            const int sr  = h + dr;
            const bool rowok = (unsigned)sr < (unsigned)HH;
            const int  src   = rowok ? sr : h;    // clamped, always valid
            const float* rp  = x + (((size_t)(b * CIN + ci) * HH + src) << 8);
            const int c  = w0 + (lane & 15) + dc;
            const int cc = c < 0 ? 0 : (c > WW - 1 ? WW - 1 : c);
            const float v = rp[cc];
            bf[i] = (__bf16)((rowok && (unsigned)c < (unsigned)WW) ? v : 0.0f);
        }

        // two WMMAs share one B fragment (doubles FLOPs per B byte)
        acc0 = __builtin_amdgcn_wmma_f32_16x16x32_bf16(
            false, a0, false, bf, (short)0, acc0, false, false);
        acc1 = __builtin_amdgcn_wmma_f32_16x16x32_bf16(
            false, a1, false, bf, (short)0, acc1, false, false);
    }

    // --- epilogue: C layout (VGPR j: M = j + 8*(lane>=16), N = lane&15) ---
    const int nn = lane & 15;
    const int mh = (lane >> 4) * 8;
    #pragma unroll
    for (int j = 0; j < 8; ++j) {
        const int co0 = mh + j;
        const int co1 = 16 + mh + j;
        out[(((size_t)(b * COUT + co0) * HH + h) << 8) + w0 + nn] = acc0[j];
        out[(((size_t)(b * COUT + co1) * HH + h) << 8) + w0 + nn] = acc1[j];
    }
}

// ---------------------------------------------------------------------------
extern "C" void kernel_launch(void* const* d_in, const int* in_sizes, int n_in,
                              void* d_out, int out_size, void* d_ws, size_t ws_size,
                              hipStream_t stream) {
    const float* x      = (const float*)d_in[0];  // [16,32,256,256]
    const float* weight = (const float*)d_in[1];  // [256,32,3,3]
    const float* fc_a_w = (const float*)d_in[2];  // [8,32]
    const float* fc_a_b = (const float*)d_in[3];  // [8]
    const float* fc_t_w = (const float*)d_in[4];  // [8,32]
    const float* fc_t_b = (const float*)d_in[5];  // [8]
    float*       out    = (float*)d_out;          // [16,32,256,256]

    float*  ws_avg  = (float*)d_ws;                   // 512 floats
    __bf16* ws_frag = (__bf16*)((char*)d_ws + 4096);  // 16*9*2*32*16 bf16

    avg_kernel<<<BB * CIN, 256, 0, stream>>>(x, ws_avg);
    wgen_kernel<<<dim3(BB, 8), 256, 0, stream>>>(ws_avg, weight, fc_a_w, fc_a_b,
                                                 fc_t_w, fc_t_b, ws_frag);
    conv_kernel<<<BB * HH * 2, 256, 0, stream>>>(x, ws_frag, out);
}
